// TransformerTransducer_42279658062526
// MI455X (gfx1250) — matrix operand order repaired and checked by reading, not compiled, not run on hardware
//
#include <hip/hip_runtime.h>
#include <hip/hip_bf16.h>
#include <cmath>

// Problem dims (fixed by reference)
#define B_  2
#define M_  384
#define N_  96
#define D_  512
#define J_  640
#define C_  1024
#define JP_ 648   // padded LDS row stride (halfs): 1296B -> bank-conflict-free

typedef __attribute__((ext_vector_type(16))) _Float16 v16h;
typedef __attribute__((ext_vector_type(8)))  _Float16 v8h;
typedef __attribute__((ext_vector_type(8)))  float    v8f;

__device__ __forceinline__ float fast_tanhf(float x) {
  x = fminf(fmaxf(x, -9.0f), 9.0f);
  float e = __expf(2.0f * x);
  return __fdividef(e - 1.0f, e + 1.0f);
}

// ---------------------------------------------------------------------------
// Phase 1: Y[r, j] = X[r, :] @ W[:, j] + bias[j]   (X: [R, 512], W: [512, 640])
// 8 rows per block, 640 threads (one column each), X tile staged in LDS.
// ---------------------------------------------------------------------------
__global__ __launch_bounds__(640) void proj_kernel(
    const float* __restrict__ X, const float* __restrict__ W,
    const float* __restrict__ bias, float* __restrict__ Y) {
  __shared__ float xs[8 * D_];
  const int r0 = blockIdx.x * 8;
  for (int i = threadIdx.x; i < 8 * D_; i += 640)
    xs[i] = X[(size_t)r0 * D_ + i];
  __syncthreads();

  const int j = threadIdx.x;  // 0..639
  const float bj = bias[j];
  float acc[8];
#pragma unroll
  for (int r = 0; r < 8; ++r) acc[r] = bj;

  for (int d = 0; d < D_; ++d) {
    const float w = W[(size_t)d * J_ + j];  // coalesced across threads
#pragma unroll
    for (int r = 0; r < 8; ++r) acc[r] = fmaf(xs[r * D_ + d], w, acc[r]);
  }
#pragma unroll
  for (int r = 0; r < 8; ++r)
    Y[(size_t)(r0 + r) * J_ + j] = acc[r];
}

// ---------------------------------------------------------------------------
// Phase 2: wjT[c, j] = (f16) w_join[j, c]   (transpose + convert, 1.25 MB)
// ---------------------------------------------------------------------------
__global__ __launch_bounds__(256) void cvt_wjoin(
    const float* __restrict__ wj, _Float16* __restrict__ wjT) {
  const int idx = blockIdx.x * 256 + threadIdx.x;  // total C_*J_, exact
  const int c = idx / J_;
  const int j = idx - c * J_;
  wjT[idx] = (_Float16)wj[(size_t)j * C_ + c];
}

// ---------------------------------------------------------------------------
// Phase 3 (fused joint): per block = (b, 4 m values, 16 n values) x all C.
//   h[rt*16+n', j] = tanh(a[b,m0+rt,j] + t[b,n0+n',j])  -> f16 in LDS
//   out tile = h_tile(64x640) @ w_joinT' + b_join       -> v_wmma f32.16x16x32.f16
// 16 waves: wave w owns col-tiles [4w,4w+4), all 4 row-tiles.
// ---------------------------------------------------------------------------
__global__ __launch_bounds__(512) void joint_kernel(
    const float* __restrict__ a, const float* __restrict__ t,
    const _Float16* __restrict__ wjT, const float* __restrict__ bjoin,
    float* __restrict__ out) {
  extern __shared__ _Float16 hs[];  // 64 rows x JP_ halfs = 82944 B

  const int bx = blockIdx.x;
  const int nb = bx % (N_ / 16);
  const int mb = (bx / (N_ / 16)) % (M_ / 4);
  const int b  = bx / ((N_ / 16) * (M_ / 4));
  const int m0 = mb * 4;
  const int n0 = nb * 16;
  const int tid = threadIdx.x;

  // ---- build h tile in LDS (f16, padded rows) ----
  const float* __restrict__ ap = a + (size_t)(b * M_ + m0) * J_;
  const float* __restrict__ tp = t + (size_t)(b * N_ + n0) * J_;
  for (int idx = tid; idx < 64 * J_; idx += 512) {
    const int j  = idx % J_;
    const int rw = idx / J_;       // 0..63
    const int np = rw & 15;        // n offset
    const int rt = rw >> 4;        // m offset
    const float v = ap[rt * J_ + j] + tp[np * J_ + j];
    hs[(rt * 16 + np) * JP_ + j] = (_Float16)fast_tanhf(v);
  }
  __syncthreads();

  const int wave = tid >> 5;
  const int lane = tid & 31;
  const int ln   = lane & 15;
  const int hi   = lane >> 4;   // half-wave select (K split per ISA layout)

  int cg[4];
#pragma unroll
  for (int ct = 0; ct < 4; ++ct) cg[ct] = (wave * 4 + ct) * 16 + ln;

  v8f acc[4][4];
  const v8f z = {0.f, 0.f, 0.f, 0.f, 0.f, 0.f, 0.f, 0.f};
#pragma unroll
  for (int rt = 0; rt < 4; ++rt)
#pragma unroll
    for (int ct = 0; ct < 4; ++ct) acc[rt][ct] = z;

  // ---- K loop: 640 / 32 = 20 steps ----
#pragma unroll 1
  for (int ks = 0; ks < J_ / 32; ++ks) {
    const int kb = ks * 32;

    // B fragments: lane ln = column cg[ct]; K {0..7,16..23} (lo) / {8..15,24..31} (hi)
    v16h Bf[4];
#pragma unroll
    for (int ct = 0; ct < 4; ++ct) {
      const _Float16* bp = wjT + (size_t)cg[ct] * J_ + kb + hi * 8;
      const v8h b0 = *(const v8h*)bp;
      const v8h b1 = *(const v8h*)(bp + 16);
      Bf[ct] = __builtin_shufflevector(
          b0, b1, 0, 1, 2, 3, 4, 5, 6, 7, 8, 9, 10, 11, 12, 13, 14, 15);
    }

#pragma unroll
    for (int rt = 0; rt < 4; ++rt) {
      // A fragment: lane ln = row; K {0..7,16..23} (lo) / {8..15,24..31} (hi)
      const _Float16* hp = hs + (rt * 16 + ln) * JP_ + kb + hi * 8;
      const v8h c0 = *(const v8h*)hp;
      const v8h c1 = *(const v8h*)(hp + 16);
      const v16h Af = __builtin_shufflevector(
          c0, c1, 0, 1, 2, 3, 4, 5, 6, 7, 8, 9, 10, 11, 12, 13, 14, 15);
#pragma unroll
      for (int ct = 0; ct < 4; ++ct)
        acc[rt][ct] = __builtin_amdgcn_wmma_f32_16x16x32_f16(
            false, Af, false, Bf[ct], (short)0, acc[rt][ct], false, false);
    }
  }

  // ---- epilogue: + b_join, store. Lane's 8 values share column c, rows n'=hi*8+v
#pragma unroll
  for (int rt = 0; rt < 4; ++rt) {
    const int m = m0 + rt;
#pragma unroll
    for (int ct = 0; ct < 4; ++ct) {
      const int c = cg[ct];
      const float bj = bjoin[c];
#pragma unroll
      for (int v = 0; v < 8; ++v) {
        const int n = n0 + hi * 8 + v;
        out[((size_t)(b * M_ + m) * N_ + n) * C_ + c] = acc[rt][ct][v] + bj;
      }
    }
  }
}

// ---------------------------------------------------------------------------
extern "C" void kernel_launch(void* const* d_in, const int* in_sizes, int n_in,
                              void* d_out, int out_size, void* d_ws, size_t ws_size,
                              hipStream_t stream) {
  const float* enc     = (const float*)d_in[0];
  const float* dec     = (const float*)d_in[1];
  const float* w_audio = (const float*)d_in[2];
  const float* b_audio = (const float*)d_in[3];
  const float* w_text  = (const float*)d_in[4];
  const float* b_text  = (const float*)d_in[5];
  const float* w_join  = (const float*)d_in[6];
  const float* b_join  = (const float*)d_in[7];
  float* out = (float*)d_out;

  // Workspace layout: a (fp32) | t (fp32) | wjT (f16, transposed)  (~3.6 MB)
  float*    a   = (float*)d_ws;                       // B*M*J
  float*    t   = a + (size_t)B_ * M_ * J_;           // B*N*J
  _Float16* wjT = (_Float16*)(t + (size_t)B_ * N_ * J_);  // C*J (32B-aligned)

  proj_kernel<<<(B_ * M_) / 8, 640, 0, stream>>>(enc, w_audio, b_audio, a);
  proj_kernel<<<(B_ * N_) / 8, 640, 0, stream>>>(dec, w_text, b_text, t);
  cvt_wjoin<<<(C_ * J_) / 256, 256, 0, stream>>>(w_join, wjT);

  const int smem = 64 * JP_ * (int)sizeof(_Float16);  // 82944 B < 320 KB/WGP
  (void)hipFuncSetAttribute(reinterpret_cast<const void*>(&joint_kernel),
                            hipFuncAttributeMaxDynamicSharedMemorySize, smem);
  joint_kernel<<<B_ * (M_ / 4) * (N_ / 16), 512, smem, stream>>>(
      a, t, wjT, b_join, out);
}
